// Attention_86801289052504
// MI455X (gfx1250) — matrix operand order repaired and checked by reading, not compiled, not run on hardware
//
#include <hip/hip_runtime.h>

// ---------------------------------------------------------------------------
// Causal self-attention, bf16 WMMA flash-attention for MI455X (gfx1250).
//   B=4, T=4096, D=U=128.  Stage 1: QKV projection (X@Wq/Wk/Wv) -> bf16 ws.
//   Stage 2: flash attention, scores scaled by 1/sqrt(T)=1/64, causal mask.
// ---------------------------------------------------------------------------

typedef __attribute__((ext_vector_type(16))) __bf16        v16bf;
typedef __attribute__((ext_vector_type(8)))  float         v8f;
typedef __attribute__((ext_vector_type(4)))  unsigned int  u32x4;

union Frag { v16bf v; u32x4 q[2]; };

#define T_SEQ  4096
#define DMODEL 128

__device__ __forceinline__ unsigned short f2bf(float f) {
  unsigned u = __float_as_uint(f);
  u += 0x7FFFu + ((u >> 16) & 1u);      // round-to-nearest-even
  return (unsigned short)(u >> 16);
}

// A-fragment (16x32 bf16): lane holds row (lane&15); elements kk+hi8..+7 and
// kk+16+hi8..+7  (ISA 7.12.2, 16-bit A layout)
__device__ __forceinline__ v16bf loadA(const unsigned short* rowbase, int kk, int hi8) {
  Frag f;
  f.q[0] = *(const u32x4*)(rowbase + kk + hi8);
  f.q[1] = *(const u32x4*)(rowbase + kk + 16 + hi8);
  return f.v;
}
// B-fragment (32x16 bf16): lane holds column (lane&15); elements kk+hi8..+7 and
// kk+16+hi8..+7 (hi8 = hi16/2), matching the A layout, from row of B^T
__device__ __forceinline__ v16bf loadB(const unsigned short* rowbase, int kk, int hi16) {
  Frag f;
  f.q[0] = *(const u32x4*)(rowbase + kk + hi16 / 2);
  f.q[1] = *(const u32x4*)(rowbase + kk + 16 + hi16 / 2);
  return f.v;
}

// ---------------------------------------------------------------------------
// Kernel 1: QKV projection.  grid = B * (T/128), block = 256 (8 waves).
// ---------------------------------------------------------------------------
__global__ __launch_bounds__(256) void proj_kernel(
    const float* __restrict__ x,
    const float* __restrict__ Wq, const float* __restrict__ Wk,
    const float* __restrict__ Wv,
    unsigned short* __restrict__ qg,     // (B,T,128) bf16 row-major
    unsigned short* __restrict__ kg,     // (B,T,128) bf16 row-major
    unsigned short* __restrict__ vtg)    // (B,128,T) bf16 (V transposed)
{
  __shared__ __align__(16) unsigned short Xl[128 * 136];   // padded stride
  __shared__ __align__(16) unsigned short Wt[128 * 136];   // W^T (u-major)

  const int tid   = threadIdx.x;
  const int lane  = tid & 31;
  const int wave  = tid >> 5;          // 0..7
  const int lanen = lane & 15;
  const int hi    = lane >> 4;

  const int b  = blockIdx.x >> 5;
  const int t0 = (blockIdx.x & 31) * 128;

  const float* xg = x + ((size_t)b * T_SEQ + t0) * DMODEL;

  // stage X tile (128x128) fp32 -> bf16, coalesced float2 reads
  for (int i = tid; i < 128 * 64; i += 256) {
    int row = i >> 6;
    int c2  = (i & 63) << 1;
    float2 f = *(const float2*)(xg + row * DMODEL + c2);
    unsigned pk = ((unsigned)f2bf(f.y) << 16) | (unsigned)f2bf(f.x);
    *(unsigned*)(&Xl[row * 136 + c2]) = pk;
  }
  __syncthreads();

  // per-wave A fragments (rows wave*16 .. +15), reused for all 24 tiles
  const unsigned short* arow = &Xl[(wave * 16 + lanen) * 136];
  v16bf af[4];
#pragma unroll
  for (int kc = 0; kc < 4; ++kc) af[kc] = loadA(arow, kc * 32, hi * 8);

  const float* Ws[3] = {Wq, Wk, Wv};
  const int m0 = wave * 16 + hi * 8;     // first output row (of 8) in tile

  for (int mat = 0; mat < 3; ++mat) {
    __syncthreads();                                   // Wt free to overwrite
    const float* wsrc = Ws[mat];
    for (int i = tid; i < 128 * 128; i += 256) {       // transpose-stage W
      int d = i >> 7, u = i & 127;
      Wt[u * 136 + d] = f2bf(wsrc[i]);
    }
    __syncthreads();

    // uniform (base,rowstep,colstep) -> branch-free store loops
    unsigned short* matbase;
    size_t rowstep, colstep;
    if (mat == 0)      { matbase = qg  + ((size_t)b * T_SEQ + t0) * DMODEL; rowstep = DMODEL; colstep = 1; }
    else if (mat == 1) { matbase = kg  + ((size_t)b * T_SEQ + t0) * DMODEL; rowstep = DMODEL; colstep = 1; }
    else               { matbase = vtg +  (size_t)b * DMODEL * T_SEQ + t0;  rowstep = 1;      colstep = T_SEQ; }

#pragma unroll
    for (int nt = 0; nt < 8; ++nt) {
      const unsigned short* brow = &Wt[(nt * 16 + lanen) * 136];
      v16bf bw[4];
#pragma unroll
      for (int kc = 0; kc < 4; ++kc) bw[kc] = loadB(brow, kc * 32, hi * 16);

      v8f acc = {0.f, 0.f, 0.f, 0.f, 0.f, 0.f, 0.f, 0.f};
#pragma unroll
      for (int kc = 0; kc < 4; ++kc)
        acc = __builtin_amdgcn_wmma_f32_16x16x32_bf16(
            false, af[kc], false, bw[kc], (short)0, acc, false, false);

      const int u = nt * 16 + lanen;
      unsigned short* dstp = matbase + (size_t)m0 * rowstep + (size_t)u * colstep;
#pragma unroll
      for (int r = 0; r < 8; ++r)                      // C layout: M=r+8*hi
        dstp[(size_t)r * rowstep] = f2bf(acc[r]);
    }
  }
}

// ---------------------------------------------------------------------------
// Kernel 2: flash attention.  grid = B * (T/64), block = 128 (4 waves).
// Each wave owns 16 query rows; block iterates causal key tiles of 64.
// ---------------------------------------------------------------------------
__global__ __launch_bounds__(128) void flash_kernel(
    const unsigned short* __restrict__ qg,
    const unsigned short* __restrict__ kg,
    const unsigned short* __restrict__ vtg,
    float* __restrict__ out)
{
  __shared__ __align__(16) unsigned short Kl[64 * 136];   // K tile, row-major
  __shared__ __align__(16) unsigned short Vl[128 * 72];   // V^T tile (u x key)
  __shared__ __align__(16) unsigned short Pl[4 * 16 * 72];// per-wave P repack

  const int tid   = threadIdx.x;
  const int lane  = tid & 31;
  const int wave  = tid >> 5;          // 0..3
  const int lanen = lane & 15;
  const int hi    = lane >> 4;

  const int b  = blockIdx.x >> 6;
  const int bq = blockIdx.x & 63;
  const int q0 = bq * 64;

  const unsigned short* qb  = qg  + (size_t)b * T_SEQ * DMODEL;
  const unsigned short* kbp = kg  + (size_t)b * T_SEQ * DMODEL;
  const unsigned short* vb  = vtg + (size_t)b * DMODEL * T_SEQ;

  // Q A-fragments straight from global (rows q0 + wave*16 + lanen)
  v16bf aq[4];
  {
    const unsigned short* qrow = qb + (size_t)(q0 + wave * 16 + lanen) * DMODEL;
#pragma unroll
    for (int kc = 0; kc < 4; ++kc) {
      Frag f;
      f.q[0] = *(const u32x4*)(qrow + kc * 32 + hi * 8);
      f.q[1] = *(const u32x4*)(qrow + kc * 32 + 16 + hi * 8);
      aq[kc] = f.v;
    }
  }

  v8f accO[8];
#pragma unroll
  for (int nt = 0; nt < 8; ++nt) {
    v8f z = {0.f, 0.f, 0.f, 0.f, 0.f, 0.f, 0.f, 0.f};
    accO[nt] = z;
  }
  float mrow[8], lrow[8];
#pragma unroll
  for (int r = 0; r < 8; ++r) { mrow[r] = -3.0e38f; lrow[r] = 0.f; }

  const float scale = 0.015625f;  // 1/sqrt(T) = 1/64 (reference scales by T!)
  const int   qrow0 = q0 + wave * 16 + hi * 8;   // first of this lane's 8 rows

  for (int j = 0; j <= bq; ++j) {
    const int kbase = j * 64;
    __syncthreads();                       // prior reads of Kl/Vl complete
    {   // stage K tile (64 x 128 bf16): thread -> (row=tid/2, half=tid&1)
      int row = tid >> 1, half = tid & 1;
      const u32x4* src = (const u32x4*)(kbp + (size_t)(kbase + row) * DMODEL + half * 64);
      u32x4* dst = (u32x4*)(&Kl[row * 136 + half * 64]);
#pragma unroll
      for (int i = 0; i < 8; ++i) dst[i] = src[i];
    }
    {   // stage V^T tile (128 x 64 bf16): thread = u row, contiguous copy
      const u32x4* src = (const u32x4*)(vb + (size_t)tid * T_SEQ + kbase);
      u32x4* dst = (u32x4*)(&Vl[tid * 72]);
#pragma unroll
      for (int i = 0; i < 8; ++i) dst[i] = src[i];
    }
    if (j < bq) {   // global_prefetch_b8 for next tiles
      __builtin_prefetch(kbp + (size_t)(kbase + 64 + (tid >> 1)) * DMODEL, 0, 1);
      __builtin_prefetch(vb + (size_t)tid * T_SEQ + kbase + 64, 0, 1);
    }
    __syncthreads();

    // ---- S = Q * K^T (16 x 64, four 16-key subtiles) ----
    const bool diag = (j == bq);
    v8f s[4];
#pragma unroll
    for (int t = 0; t < 4; ++t) {
      const unsigned short* krow = &Kl[(t * 16 + lanen) * 136];
      v16bf bk[4];
#pragma unroll
      for (int kc = 0; kc < 4; ++kc) bk[kc] = loadB(krow, kc * 32, hi * 16);

      v8f acc = {0.f, 0.f, 0.f, 0.f, 0.f, 0.f, 0.f, 0.f};
#pragma unroll
      for (int kc = 0; kc < 4; ++kc)
        acc = __builtin_amdgcn_wmma_f32_16x16x32_bf16(
            false, aq[kc], false, bk[kc], (short)0, acc, false, false);

      // scale + causal mask, branch-free (v_cndmask)
      const int key = kbase + t * 16 + lanen;
#pragma unroll
      for (int r = 0; r < 8; ++r) {
        float v = acc[r] * scale;
        acc[r] = (diag && key > qrow0 + r) ? -1.0e30f : v;
      }
      s[t] = acc;
    }

    // ---- online softmax (row stats across 16-lane half via shfl_xor) ----
    float mnew[8], alpha[8];
#pragma unroll
    for (int r = 0; r < 8; ++r) {
      float m = fmaxf(fmaxf(s[0][r], s[1][r]), fmaxf(s[2][r], s[3][r]));
#pragma unroll
      for (int d = 1; d < 16; d <<= 1) m = fmaxf(m, __shfl_xor(m, d, 32));
      mnew[r]  = fmaxf(mrow[r], m);
      alpha[r] = __expf(mrow[r] - mnew[r]);
      mrow[r]  = mnew[r];
    }

    unsigned short* pl = &Pl[wave * 16 * 72];
    float rsum[8];
#pragma unroll
    for (int r = 0; r < 8; ++r) rsum[r] = 0.f;
#pragma unroll
    for (int t = 0; t < 4; ++t)
#pragma unroll
      for (int r = 0; r < 8; ++r) {
        float p = __expf(s[t][r] - mnew[r]);
        rsum[r] += p;
        // write in C layout position -> Pl becomes row-major 16x64 strip
        pl[(hi * 8 + r) * 72 + t * 16 + lanen] = f2bf(p);
      }
#pragma unroll
    for (int r = 0; r < 8; ++r) {
      float rs = rsum[r];
#pragma unroll
      for (int d = 1; d < 16; d <<= 1) rs += __shfl_xor(rs, d, 32);
      lrow[r] = lrow[r] * alpha[r] + rs;
    }

    // rescale O accumulators
#pragma unroll
    for (int nt = 0; nt < 8; ++nt)
#pragma unroll
      for (int r = 0; r < 8; ++r) accO[nt][r] *= alpha[r];

    // re-read P as A-fragments (same-wave DS ops are in order)
    v16bf ap[2];
    const unsigned short* prow = &Pl[wave * 16 * 72 + lanen * 72];
#pragma unroll
    for (int kc = 0; kc < 2; ++kc) {
      Frag f;
      f.q[0] = *(const u32x4*)(prow + kc * 32 + hi * 8);
      f.q[1] = *(const u32x4*)(prow + kc * 32 + 16 + hi * 8);
      ap[kc] = f.v;
    }

    // ---- O += P * V ----
#pragma unroll
    for (int nt = 0; nt < 8; ++nt) {
      const unsigned short* vrow = &Vl[(nt * 16 + lanen) * 72];
      v16bf bv[2];
#pragma unroll
      for (int kc = 0; kc < 2; ++kc) bv[kc] = loadB(vrow, kc * 32, hi * 16);

      v8f acc = accO[nt];
#pragma unroll
      for (int kc = 0; kc < 2; ++kc)
        acc = __builtin_amdgcn_wmma_f32_16x16x32_bf16(
            false, ap[kc], false, bv[kc], (short)0, acc, false, false);
      accO[nt] = acc;
    }
  }

  // epilogue: divide by row sums, store fp32
#pragma unroll
  for (int r = 0; r < 8; ++r) lrow[r] = 1.f / lrow[r];
  float* outb = out + ((size_t)b * T_SEQ + q0 + wave * 16 + hi * 8) * DMODEL;
#pragma unroll
  for (int nt = 0; nt < 8; ++nt) {
    const int u = nt * 16 + lanen;
#pragma unroll
    for (int r = 0; r < 8; ++r)
      outb[(size_t)r * DMODEL + u] = accO[nt][r] * lrow[r];
  }
}

// ---------------------------------------------------------------------------
extern "C" void kernel_launch(void* const* d_in, const int* in_sizes, int n_in,
                              void* d_out, int out_size, void* d_ws, size_t ws_size,
                              hipStream_t stream) {
  (void)in_sizes; (void)n_in; (void)out_size; (void)ws_size;
  const float* x  = (const float*)d_in[0];
  const float* Wq = (const float*)d_in[1];
  const float* Wk = (const float*)d_in[2];
  const float* Wv = (const float*)d_in[3];
  float* out = (float*)d_out;

  const size_t elems = (size_t)4 * T_SEQ * DMODEL;        // 2,097,152 per tensor
  unsigned short* qg = (unsigned short*)d_ws;             // bf16 Q
  unsigned short* kg = qg + elems;                        // bf16 K
  unsigned short* vt = kg + elems;                        // bf16 V^T  (needs 12 MB ws)

  proj_kernel<<<4 * (T_SEQ / 128), 256, 0, stream>>>(x, Wq, Wk, Wv, qg, kg, vt);
  flash_kernel<<<4 * (T_SEQ / 64), 128, 0, stream>>>(qg, kg, vt, out);
}
